// GAT_91319594647659
// MI455X (gfx1250) — matrix operand order repaired and checked by reading, not compiled, not run on hardware
//
#include <hip/hip_runtime.h>

// ---------------- problem constants (from reference) ----------------
#define N_NODES 100000
#define N_EDGES 1600000
#define IN_F    128
#define MID     256          // H * HID_F  == H * OUT_F
#define HEADS   4
#define DHEAD   64
#define NEG_SLOPE 0.2f

typedef __attribute__((ext_vector_type(16))) __bf16 v16bf;
typedef __attribute__((ext_vector_type(8)))  float  v8f;

// ---------------- helpers ----------------
static __device__ __forceinline__ unsigned short f2bf_rne(float f) {
    unsigned u = __float_as_uint(f);
    u += 0x7FFFu + ((u >> 16) & 1u);       // round-to-nearest-even
    return (unsigned short)(u >> 16);
}
// order-preserving float<->uint mapping (for atomic segment-max)
static __device__ __forceinline__ unsigned keyenc(float f) {
    unsigned b = __float_as_uint(f);
    return b ^ ((b & 0x80000000u) ? 0xFFFFFFFFu : 0x80000000u);
}
static __device__ __forceinline__ float keydec(unsigned k) {
    unsigned b = k ^ ((k & 0x80000000u) ? 0x80000000u : 0xFFFFFFFFu);
    return __uint_as_float(b);
}

// ---------------- f32 -> bf16 buffer convert ----------------
__global__ void k_cvt_bf16(const float* __restrict__ in,
                           unsigned short* __restrict__ out, int n) {
    int i = blockIdx.x * blockDim.x + threadIdx.x;
    if (i < n) out[i] = f2bf_rne(in[i]);
}

// ---------------- WMMA GEMM: C[N,256] = X[N,K] * W[256,K]^T ----------------
// One wave per 16-row strip of nodes. A-frags for all K-tiles held in regs.
// KT = K/32 (4 for layer1, 8 for layer2). M fixed at 256 (16 n-tiles).
template <int KT>
__global__ __launch_bounds__(256)
void k_gemm_wmma(const unsigned short* __restrict__ Xb,   // [N, K] bf16 bits
                 const unsigned short* __restrict__ Wb,   // [256, K] bf16 bits
                 float* __restrict__ outF,                // [N, 256] f32
                 int nstrips) {
    const int K    = KT * 32;
    const int wave = (int)((blockIdx.x * blockDim.x + threadIdx.x) >> 5);
    const int lane = (int)(threadIdx.x & 31);
    if (wave >= nstrips) return;                // uniform per wave: EXEC stays full

    const int m  = lane & 15;                   // row-in-tile / col-in-tile
    const int hi = lane >> 4;                   // lane half

    union AU { uint4 u[2]; v16bf v; };
    // A fragment per k-tile: lane holds row m, K = hi*8 + [0..7] and +16..+23
    AU a[KT];
    const unsigned short* arow = Xb + (size_t)(wave * 16 + m) * K + hi * 8;
#pragma unroll
    for (int kt = 0; kt < KT; ++kt) {
        const unsigned short* p = arow + kt * 32;
        a[kt].u[0] = *(const uint4*)(p);
        a[kt].u[1] = *(const uint4*)(p + 16);
    }

    const unsigned short* bbase = Wb + (size_t)m * K + hi * 8;
#pragma unroll
    for (int nt = 0; nt < 16; ++nt) {
        v8f acc = {};
        const unsigned short* bc = bbase + (size_t)nt * 16 * K;
#pragma unroll
        for (int kt = 0; kt < KT; ++kt) {
            AU b;                               // lane: col m, K = hi*8 + [0..7] and +16..+23
            const unsigned short* p = bc + kt * 32;
            b.u[0] = *(const uint4*)(p);
            b.u[1] = *(const uint4*)(p + 16);
            acc = __builtin_amdgcn_wmma_f32_16x16x32_bf16(
                false, a[kt].v, false, b.v, (short)0, acc, false, false);
        }
        // D layout: VGPR r -> (M = r + hi*8, N = m)
        float* o = outF + (size_t)(wave * 16 + hi * 8) * 256 + nt * 16 + m;
#pragma unroll
        for (int r = 0; r < 8; ++r) o[(size_t)r * 256] = acc[r];
    }
}

// ---------------- per-node attention logits: el/er [N,H] ----------------
__global__ void k_node_attn(const float* __restrict__ feat,
                            const float* __restrict__ al,
                            const float* __restrict__ ar,
                            float* __restrict__ el, float* __restrict__ er) {
    int idx = blockIdx.x * blockDim.x + threadIdx.x;
    if (idx >= N_NODES * HEADS) return;
    int n = idx >> 2, h = idx & 3;
    const float4* f = (const float4*)(feat + (size_t)n * MID + h * DHEAD);
    const float4* A = (const float4*)(al + h * DHEAD);
    const float4* B = (const float4*)(ar + h * DHEAD);
    float sl = 0.f, sr = 0.f;
#pragma unroll
    for (int j = 0; j < DHEAD / 4; ++j) {
        float4 v = f[j], x = A[j], y = B[j];
        sl += v.x * x.x + v.y * x.y + v.z * x.z + v.w * x.w;
        sr += v.x * y.x + v.y * y.y + v.z * y.z + v.w * y.w;
    }
    el[idx] = sl; er[idx] = sr;
}

// ---------------- edge logits + segment max ----------------
__global__ void k_edge_logits(const int* __restrict__ src, const int* __restrict__ dst,
                              const float* __restrict__ el, const float* __restrict__ er,
                              float* __restrict__ eval, unsigned* __restrict__ emax) {
    int idx = blockIdx.x * blockDim.x + threadIdx.x;
    if (idx >= N_EDGES * HEADS) return;
    int e = idx >> 2, h = idx & 3;
    int s = src[e], d = dst[e];
    float v = el[s * HEADS + h] + er[d * HEADS + h];
    v = (v > 0.f) ? v : NEG_SLOPE * v;          // leaky relu
    eval[idx] = v;
    atomicMax(&emax[d * HEADS + h], keyenc(v));
}

// ---------------- edge exp + segment sum ----------------
__global__ void k_edge_exp(const int* __restrict__ dst,
                           const unsigned* __restrict__ emax,
                           float* __restrict__ eval, float* __restrict__ denom) {
    int idx = blockIdx.x * blockDim.x + threadIdx.x;
    if (idx >= N_EDGES * HEADS) return;
    int e = idx >> 2, h = idx & 3;
    int d = dst[e];
    float mx = keydec(emax[d * HEADS + h]);
    float a = __expf(eval[idx] - mx);
    eval[idx] = a;
    atomicAdd(&denom[d * HEADS + h], a);
}

// ---------------- message pass: one wave per edge ----------------
__global__ __launch_bounds__(256)
void k_edge_msg(const int* __restrict__ src, const int* __restrict__ dst,
                const float* __restrict__ feat, const float* __restrict__ eval,
                const float* __restrict__ denom, float* __restrict__ agg) {
    long long gid = (long long)blockIdx.x * blockDim.x + threadIdx.x;
    int e    = (int)(gid >> 5);
    int lane = (int)(gid & 31);
    if (e >= N_EDGES) return;
    int s = src[e], d = dst[e];
    int h = lane >> 3;                                   // lane*8 spans one head block
    float c = eval[e * HEADS + h] / denom[d * HEADS + h];
    const float4* fs = (const float4*)(feat + (size_t)s * MID + lane * 8);
    float4 v0 = fs[0], v1 = fs[1];
    float* od = agg + (size_t)d * MID + lane * 8;
    atomicAdd(od + 0, v0.x * c); atomicAdd(od + 1, v0.y * c);
    atomicAdd(od + 2, v0.z * c); atomicAdd(od + 3, v0.w * c);
    atomicAdd(od + 4, v1.x * c); atomicAdd(od + 5, v1.y * c);
    atomicAdd(od + 6, v1.z * c); atomicAdd(od + 7, v1.w * c);
}

// ---------------- bias + ELU + recast to bf16 (input of layer-2 GEMM) ----------------
__global__ void k_elu_bf16(const float* __restrict__ agg, const float* __restrict__ bias,
                           unsigned short* __restrict__ outb) {
    int idx = blockIdx.x * blockDim.x + threadIdx.x;
    if (idx >= N_NODES * MID) return;
    float x = agg[idx] + bias[idx & (MID - 1)];
    float y = (x > 0.f) ? x : (__expf(x) - 1.f);
    outb[idx] = f2bf_rne(y);
}

// ---------------- final: bias + mean over heads -> [N,64] ----------------
__global__ void k_final(const float* __restrict__ agg, const float* __restrict__ bias,
                        float* __restrict__ out) {
    int idx = blockIdx.x * blockDim.x + threadIdx.x;
    if (idx >= N_NODES * DHEAD) return;
    int n = idx >> 6, f = idx & 63;
    const float* a = agg + (size_t)n * MID;
    float s = 0.f;
#pragma unroll
    for (int h = 0; h < HEADS; ++h) s += a[h * DHEAD + f] + bias[h * DHEAD + f];
    out[idx] = 0.25f * s;
}

// =====================================================================
extern "C" void kernel_launch(void* const* d_in, const int* in_sizes, int n_in,
                              void* d_out, int out_size, void* d_ws, size_t ws_size,
                              hipStream_t stream) {
    const float* x   = (const float*)d_in[0];
    const int*   src = (const int*)  d_in[1];
    const int*   dst = (const int*)  d_in[2];
    const float* W1  = (const float*)d_in[3];
    const float* al1 = (const float*)d_in[4];
    const float* ar1 = (const float*)d_in[5];
    const float* b1  = (const float*)d_in[6];
    const float* W2  = (const float*)d_in[7];
    const float* al2 = (const float*)d_in[8];
    const float* ar2 = (const float*)d_in[9];
    const float* b2  = (const float*)d_in[10];
    float* out = (float*)d_out;
    (void)in_sizes; (void)n_in; (void)out_size; (void)ws_size;

    // -------- workspace carve-up (~288 MB, 256B aligned) --------
    char* w = (char*)d_ws;
    auto carve = [&](size_t bytes) { char* p = w; w += (bytes + 255) & ~(size_t)255; return p; };
    unsigned short* Xb    = (unsigned short*)carve((size_t)N_NODES * MID * 2);   // x_bf16 / h_bf16
    unsigned short* Wb    = (unsigned short*)carve((size_t)MID * MID * 2);       // W1b / W2b
    float*          feat  = (float*)carve((size_t)N_NODES * MID * 4);            // projected feats
    float*          agg   = (float*)carve((size_t)N_NODES * MID * 4);            // segment sums
    float*          el    = (float*)carve((size_t)N_NODES * HEADS * 4);
    float*          er    = (float*)carve((size_t)N_NODES * HEADS * 4);
    unsigned*       emax  = (unsigned*)carve((size_t)N_NODES * HEADS * 4);
    float*          denom = (float*)carve((size_t)N_NODES * HEADS * 4);
    float*          eval  = (float*)carve((size_t)N_EDGES * HEADS * 4);

    const int NSTRIPS = N_NODES / 16;            // 6250 exactly
    const dim3 B256(256);
    auto blocks = [](long long n) { return dim3((unsigned)((n + 255) / 256)); };

    // ================= layer 1 =================
    k_cvt_bf16<<<blocks((long long)N_NODES * IN_F), B256, 0, stream>>>(x, Xb, N_NODES * IN_F);
    k_cvt_bf16<<<blocks(MID * IN_F), B256, 0, stream>>>(W1, Wb, MID * IN_F);
    k_gemm_wmma<IN_F / 32><<<dim3((NSTRIPS + 7) / 8), B256, 0, stream>>>(Xb, Wb, feat, NSTRIPS);

    k_node_attn<<<blocks(N_NODES * HEADS), B256, 0, stream>>>(feat, al1, ar1, el, er);
    hipMemsetAsync(emax,  0, (size_t)N_NODES * HEADS * 4, stream);
    hipMemsetAsync(denom, 0, (size_t)N_NODES * HEADS * 4, stream);
    hipMemsetAsync(agg,   0, (size_t)N_NODES * MID * 4, stream);
    k_edge_logits<<<blocks((long long)N_EDGES * HEADS), B256, 0, stream>>>(src, dst, el, er, eval, emax);
    k_edge_exp   <<<blocks((long long)N_EDGES * HEADS), B256, 0, stream>>>(dst, emax, eval, denom);
    k_edge_msg   <<<blocks((long long)N_EDGES * 32),    B256, 0, stream>>>(src, dst, feat, eval, denom, agg);

    k_elu_bf16<<<blocks((long long)N_NODES * MID), B256, 0, stream>>>(agg, b1, Xb);

    // ================= layer 2 =================
    k_cvt_bf16<<<blocks(MID * MID), B256, 0, stream>>>(W2, Wb, MID * MID);
    k_gemm_wmma<MID / 32><<<dim3((NSTRIPS + 7) / 8), B256, 0, stream>>>(Xb, Wb, feat, NSTRIPS);

    k_node_attn<<<blocks(N_NODES * HEADS), B256, 0, stream>>>(feat, al2, ar2, el, er);
    hipMemsetAsync(emax,  0, (size_t)N_NODES * HEADS * 4, stream);
    hipMemsetAsync(denom, 0, (size_t)N_NODES * HEADS * 4, stream);
    hipMemsetAsync(agg,   0, (size_t)N_NODES * MID * 4, stream);
    k_edge_logits<<<blocks((long long)N_EDGES * HEADS), B256, 0, stream>>>(src, dst, el, er, eval, emax);
    k_edge_exp   <<<blocks((long long)N_EDGES * HEADS), B256, 0, stream>>>(dst, emax, eval, denom);
    k_edge_msg   <<<blocks((long long)N_EDGES * 32),    B256, 0, stream>>>(src, dst, feat, eval, denom, agg);

    k_final<<<blocks((long long)N_NODES * DHEAD), B256, 0, stream>>>(agg, b2, out);
}
